// SimpleNetworkAD_12197707121112
// MI455X (gfx1250) — matrix operand order repaired and checked by reading, not compiled, not run on hardware
//
#include <hip/hip_runtime.h>

typedef __attribute__((ext_vector_type(16))) _Float16 v16h;
typedef __attribute__((ext_vector_type(8)))  float    v8f;

#define NODE 480
#define H    20
#define BATCH 16384
#define WAVES 8
#define BLOCKS_Y 8
#define ROWS_PER_BLOCK (BATCH / BLOCKS_Y)      /* 2048 */
#define TILES_PER_BLOCK (ROWS_PER_BLOCK / 16)  /* 128  */
#define PAIRS_PER_BLOCK (TILES_PER_BLOCK / 2)  /* 64   */
#define STRIDE 36  /* floats per staged row: 144B, 16B-aligned, bank-friendly */

__device__ __forceinline__ float fast_tanh(float x) {
#if __has_builtin(__builtin_amdgcn_tanhf)
  // Native V_TANH_F32: single TRANS-pipe op, co-executes with XDL WMMA.
  return __builtin_amdgcn_tanhf(x);
#elif __has_builtin(__builtin_amdgcn_tanh_f32)
  return __builtin_amdgcn_tanh_f32(x);
#else
  float e = __builtin_amdgcn_exp2f(x * 2.88539008177792681472f);
  return 1.0f - 2.0f * __builtin_amdgcn_rcpf(e + 1.0f);
#endif
}

__global__ __launch_bounds__(256, 1)
void node_mlp_kernel(const float* __restrict__ x,   // [B, NODE]
                     const float* __restrict__ W1,  // [NODE, H]
                     const float* __restrict__ b1,  // [NODE, H]
                     const float* __restrict__ W2,  // [NODE, H, H]
                     const float* __restrict__ b2,  // [NODE, H]
                     const float* __restrict__ W3,  // [NODE, H, H]
                     const float* __restrict__ b3,  // [NODE, H]
                     const float* __restrict__ W4,  // [NODE, H]
                     const float* __restrict__ b4,  // [NODE]
                     float* __restrict__ out)       // [B, NODE]
{
  // two staging buffers per wave: 2-tile software pipeline
  __shared__ float stage[WAVES][2][16 * STRIDE];

  const int g     = blockIdx.x;                    // node id
  const int ybase = blockIdx.y * ROWS_PER_BLOCK;
  const int wave  = threadIdx.x >> 5;
  const int lane  = threadIdx.x & 31;
  const int laneN = lane & 15;
  const int hi    = lane >> 4;

  // ---- per-lane layer-1 weights in A-fragment K order --------------------
  // 16-bit A 16x32 layout: lanes 0-15 hold K={0..7,16..23}, lanes 16-31
  // hold K={8..15,24..31}; element j -> k = (j<8 ? j : j+8) + 8*hi.
  float w1k[16], b1k[16];
#pragma unroll
  for (int j = 0; j < 16; ++j) {
    int  k  = (j < 8 ? j : j + 8) + hi * 8;
    bool ok = (k < H);
    w1k[j] = ok ? W1[g * H + k] : 0.0f;
    b1k[j] = ok ? b1[g * H + k] : 0.0f;
  }

  // ---- register-resident B fragments for layers 2 & 3 (zero padded) ------
  // 16-bit B 32x16 layout: lanes 0-15 hold K={0..7,16..23}, lanes 16-31
  // K={8..15,24..31}; element idx -> k = (idx<8 ? idx : idx+8) + 8*hi; column n = lane&15 (+16 for hi N-tile).
  v16h bf2[2], bf3[2];
#pragma unroll
  for (int t = 0; t < 2; ++t) {
    int n = t * 16 + laneN;
#pragma unroll
    for (int idx = 0; idx < 16; ++idx) {
      int  k  = (idx < 8 ? idx : idx + 8) + hi * 8;
      bool ok = (k < H) && (n < H);
      bf2[t][idx] = ok ? (_Float16)W2[(g * H + k) * H + n] : (_Float16)0.0f;
      bf3[t][idx] = ok ? (_Float16)W3[(g * H + k) * H + n] : (_Float16)0.0f;
    }
  }

  // Biases / output weights in D-layout lane positions (N = lane&15).
  const float bias2a = (laneN < H)      ? b2[g * H + laneN]      : 0.0f;
  const float bias2b = (16 + laneN < H) ? b2[g * H + 16 + laneN] : 0.0f;
  const float bias3a = (laneN < H)      ? b3[g * H + laneN]      : 0.0f;
  const float bias3b = (16 + laneN < H) ? b3[g * H + 16 + laneN] : 0.0f;
  const float w4a    = (laneN < H)      ? W4[g * H + laneN]      : 0.0f;
  const float w4b    = (16 + laneN < H) ? W4[g * H + 16 + laneN] : 0.0f;
  const float bias4  = b4[g];

  for (int tp = wave; tp < PAIRS_PER_BLOCK; tp += WAVES) {
    int r0[2];
    r0[0] = ybase + (2 * tp + 0) * 16;
    r0[1] = ybase + (2 * tp + 1) * 16;

    // ---- load both tiles' x columns, prefetch next pair -----------------
    float xv[2];
#pragma unroll
    for (int u = 0; u < 2; ++u)
      xv[u] = x[(size_t)(r0[u] + laneN) * NODE + g];
    if (tp + WAVES < PAIRS_PER_BLOCK) {
      __builtin_prefetch(&x[(size_t)(r0[0] + WAVES * 32 + laneN) * NODE + g], 0, 3);
    }

    // ---- layer 1 (elementwise) directly into A-fragment layout ----------
    v16h a1[2];
#pragma unroll
    for (int u = 0; u < 2; ++u)
#pragma unroll
      for (int j = 0; j < 16; ++j)
        a1[u][j] = (_Float16)fast_tanh(xv[u] * w1k[j] + b1k[j]);

    // ---- layer 2: 4 independent WMMAs (2 tiles x 2 N-tiles) -------------
    v8f z = {};
    v8f acc[4];
#pragma unroll
    for (int u = 0; u < 2; ++u) {
      acc[2 * u + 0] = __builtin_amdgcn_wmma_f32_16x16x32_f16(
          false, a1[u], false, bf2[0], (short)0, z, false, false);
      acc[2 * u + 1] = __builtin_amdgcn_wmma_f32_16x16x32_f16(
          false, a1[u], false, bf2[1], (short)0, z, false, false);
    }

    // bias + tanh, stage row-major f32 [m][k]; both tiles' stores batched so
    // tile-B VALU fills tile-A's WMMA->VALU hazard slots.
#pragma unroll
    for (int u = 0; u < 2; ++u) {
      float* st = stage[wave][u];
#pragma unroll
      for (int i = 0; i < 8; ++i) {
        int m = i + hi * 8;                     // D layout: lanes>=16 hold M=i+8
        st[m * STRIDE + laneN]      = fast_tanh(acc[2 * u + 0][i] + bias2a);
        st[m * STRIDE + 16 + laneN] = fast_tanh(acc[2 * u + 1][i] + bias2b);
      }
    }
    asm volatile("s_wait_dscnt 0" ::: "memory"); // per-wave LDS RAW fence

    // reload as A fragments: row m=laneN, K set {base..+7, base+16..+23}
    v16h a3[2];
#pragma unroll
    for (int u = 0; u < 2; ++u) {
      const float4* rp =
          (const float4*)(stage[wave][u] + laneN * STRIDE + hi * 8);
      float4 f0 = rp[0], f1 = rp[1];   // k = base .. base+7
      float4 f2 = rp[4], f3 = rp[5];   // k = base+16 .. base+23
      float tmp[16] = {f0.x, f0.y, f0.z, f0.w, f1.x, f1.y, f1.z, f1.w,
                       f2.x, f2.y, f2.z, f2.w, f3.x, f3.y, f3.z, f3.w};
#pragma unroll
      for (int j = 0; j < 16; ++j) a3[u][j] = (_Float16)tmp[j];
    }

    // ---- layer 3: 4 independent WMMAs -----------------------------------
#pragma unroll
    for (int u = 0; u < 2; ++u) {
      acc[2 * u + 0] = __builtin_amdgcn_wmma_f32_16x16x32_f16(
          false, a3[u], false, bf3[0], (short)0, z, false, false);
      acc[2 * u + 1] = __builtin_amdgcn_wmma_f32_16x16x32_f16(
          false, a3[u], false, bf3[1], (short)0, z, false, false);
    }

    // ---- layer 4: dot(h3[m,:], W4) via cross-lane reduce over N=lane&15 --
#pragma unroll
    for (int u = 0; u < 2; ++u) {
#pragma unroll
      for (int i = 0; i < 8; ++i) {
        float h3a = fast_tanh(acc[2 * u + 0][i] + bias3a);
        float h3b = fast_tanh(acc[2 * u + 1][i] + bias3b);
        float p = h3a * w4a + h3b * w4b;         // each lane: 2 K-terms
        p += __shfl_xor(p, 1, 32);               // reduce within 16-lane half
        p += __shfl_xor(p, 2, 32);
        p += __shfl_xor(p, 4, 32);
        p += __shfl_xor(p, 8, 32);
        if (laneN == 0) {
          int m = i + hi * 8;
          out[(size_t)(r0[u] + m) * NODE + g] = p + bias4;
        }
      }
    }
  }
}

extern "C" void kernel_launch(void* const* d_in, const int* in_sizes, int n_in,
                              void* d_out, int out_size, void* d_ws, size_t ws_size,
                              hipStream_t stream) {
  const float* x  = (const float*)d_in[0];
  const float* W1 = (const float*)d_in[1];
  const float* b1 = (const float*)d_in[2];
  const float* W2 = (const float*)d_in[3];
  const float* b2 = (const float*)d_in[4];
  const float* W3 = (const float*)d_in[5];
  const float* b3 = (const float*)d_in[6];
  const float* W4 = (const float*)d_in[7];
  const float* b4 = (const float*)d_in[8];
  float* out = (float*)d_out;

  dim3 grid(NODE, BLOCKS_Y, 1);   // 480 nodes x 8 batch slices = 3840 WGs
  dim3 block(256, 1, 1);          // 8 waves of 32
  node_mlp_kernel<<<grid, block, 0, stream>>>(x, W1, b1, W2, b2, W3, b3, W4, b4, out);
}
